// Net_47536698032286
// MI455X (gfx1250) — compile-verified, matrix operand order checked by reading, not run on hardware
//
#include <hip/hip_runtime.h>
#include <hip/hip_bf16.h>

// ---------------- problem constants ----------------
constexpr int Mn = 10000;        // nodes
constexpr int Kd = 768;          // input feature dim
constexpr int Hh = 12;           // heads
constexpr int Cc = 768;          // per-head dim
constexpr int Nc = Hh * Cc;      // 9216
constexpr int Eg = 20000;        // edges (without self loops)
constexpr int Et = Eg + Mn;      // edges incl. self loops
constexpr int Gg = 64;           // graphs
constexpr int NTK = Kd / 32;     // 24 K-tiles
constexpr int NTN = Nc / 64;     // 144 N-tiles

typedef __attribute__((ext_vector_type(16))) __bf16 v16bf;
typedef __attribute__((ext_vector_type(8)))  float  v8f;
typedef int v4i_vs __attribute__((vector_size(16)));   // matches builtin proto

// ---------------- async copy to LDS (CDNA5 ASYNCcnt path) ----------------
#if defined(__gfx1250__) && __has_builtin(__builtin_amdgcn_global_load_async_to_lds_b128)
#define HAVE_ASYNC_LDS 1
__device__ __forceinline__ void copy16_async(const void* g, void* l) {
  __builtin_amdgcn_global_load_async_to_lds_b128(
      (__attribute__((address_space(1))) v4i_vs*)g,
      (__attribute__((address_space(3))) v4i_vs*)l, 0, 0);
}
#else
#define HAVE_ASYNC_LDS 0
__device__ __forceinline__ void copy16_async(const void* g, void* l) {
  *(uint4*)l = *(const uint4*)g;   // synchronous fallback
}
#endif

template <int N>
__device__ __forceinline__ void wait_async() {
#if HAVE_ASYNC_LDS
#if __has_builtin(__builtin_amdgcn_s_wait_asynccnt)
  __builtin_amdgcn_s_wait_asynccnt(N);
#else
  asm volatile("s_wait_asynccnt %0" ::"i"(N) : "memory");
#endif
#endif
}

// ---------------- helpers ----------------
__global__ void zero_u32(unsigned* __restrict__ p, size_t n) {
  size_t i = (size_t)blockIdx.x * blockDim.x + threadIdx.x;
  size_t stride = (size_t)gridDim.x * blockDim.x;
  for (; i < n; i += stride) p[i] = 0u;
}

__global__ void f32_to_bf16(const float* __restrict__ in,
                            __hip_bfloat16* __restrict__ out, int n) {
  int i = blockIdx.x * blockDim.x + threadIdx.x;
  int stride = gridDim.x * blockDim.x;
  for (; i < n; i += stride) out[i] = __float2bfloat16(in[i]);
}

// monotonic key so unsigned atomicMax orders floats correctly
__device__ __forceinline__ unsigned fkey(float f) {
  unsigned u = __float_as_uint(f);
  return (u >> 31) ? ~u : (u | 0x80000000u);
}
__device__ __forceinline__ float fdec(unsigned u) {
  return __uint_as_float((u >> 31) ? (u & 0x7FFFFFFFu) : ~u);
}

// ---- W f32 [768,9216] -> bf16 pre-swizzled into WMMA fragment-tile order ----
// layout: [tk(24)][tn(144)][stn(4)][lane(32)*16 + idx], idx=(k&7)+8*(k>>4),
// lane=(n%16)+16*((k>>3)&1).  Each thread writes 8 contiguous bf16 (16B) and
// gathers 8 K-strided f32 (coalesced across lanes in N).
__global__ void w_to_bf16_swz(const float* __restrict__ W,
                              __hip_bfloat16* __restrict__ out) {
  int t = blockIdx.x * blockDim.x + threadIdx.x;
  constexpr int total = (Kd * Nc) / 8;
  if (t >= total) return;
  int idxgrp = t & 1;               // idx 0..7 vs 8..15
  int lane   = (t >> 1) & 31;
  int stn    = (t >> 6) & 3;
  int tn     = (t >> 8) % NTN;
  int tk     = (t >> 8) / NTN;
  int h      = lane >> 4;
  int n      = tn * 64 + stn * 16 + (lane & 15);
  int kbase  = tk * 32 + idxgrp * 16 + h * 8;
  __hip_bfloat16 tmp[8];
#pragma unroll
  for (int j = 0; j < 8; ++j)
    tmp[j] = __float2bfloat16(W[(size_t)(kbase + j) * Nc + n]);
  *(uint4*)(out + (size_t)t * 8) = *(const uint4*)tmp;
}

// ---------------- bf16 WMMA GEMM: D[M,Nc] = A[M,Kd] * Bswz ----------------
// block: 256 thr = 8 waves; tile 128(M) x 64(N); K-step 32; double-buffered
// async staging: issue tile t+1 (GLOBAL_LOAD_ASYNC_TO_LDS_B128) while WMMAs
// consume tile t; s_wait_asynccnt<=3 (3 async instrs/wave/stage, in-order).
__global__ __launch_bounds__(256) void gemm_bf16_wmma(
    const __hip_bfloat16* __restrict__ A,
    const __hip_bfloat16* __restrict__ Bswz,
    __hip_bfloat16* __restrict__ D) {
  __shared__ __align__(16) unsigned short lA[2][8 * 512];
  __shared__ __align__(16) unsigned short lB[2][4 * 512];

  const int tid  = threadIdx.x;
  const int wave = tid >> 5;
  const int lane = tid & 31;
  const int tM = blockIdx.x * 128;
  const int tN = blockIdx.y;             // tile index (64 cols)
  const bool full = (tM + 128 <= Mn);

  v8f acc[4] = {};

  auto stage = [&](int kt, int p) {
    if (full) {
      // A: 2 chunks of 16B per thread; fragment-contiguous destination
#pragma unroll
      for (int c = 0; c < 2; ++c) {
        int cid  = tid * 2 + c;          // 0..511
        int row  = cid >> 2;             // 0..127
        int kc0  = (cid & 3) << 3;       // 0,8,16,24
        int st   = row >> 4, mrow = row & 15;
        int ln   = mrow + (((kc0 >> 3) & 1) << 4);
        int idx0 = (kc0 >> 4) << 3;
        copy16_async(A + (size_t)(tM + row) * Kd + kt + kc0,
                     &lA[p][st * 512 + ln * 16 + idx0]);
      }
    } else {
      // last M-block: synchronous staging with zero fill
#pragma unroll
      for (int c = 0; c < 2; ++c) {
        int cid  = tid * 2 + c;
        int row  = cid >> 2;
        int kc0  = (cid & 3) << 3;
        int gm   = tM + row;
        union { uint4 q; unsigned short u[8]; } buf;
        if (gm < Mn) buf.q = *(const uint4*)(A + (size_t)gm * Kd + kt + kc0);
        else         buf.q = make_uint4(0u, 0u, 0u, 0u);
        int st = row >> 4, mrow = row & 15;
        int ln   = mrow + (((kc0 >> 3) & 1) << 4);
        int idx0 = (kc0 >> 4) << 3;
        *(uint4*)&lA[p][st * 512 + ln * 16 + idx0] = buf.q;
      }
    }
    // B: pre-swizzled, pure contiguous copy (one 16B chunk per thread)
    const __hip_bfloat16* bsrc =
        Bswz + ((size_t)(kt >> 5) * NTN + tN) * 2048 + tid * 8;
    copy16_async(bsrc, &lB[p][tid * 8]);
  };

  stage(0, 0);
  int p = 0;
  for (int kt = 0; kt < Kd; kt += 32) {
    const bool hasNext = (kt + 32) < Kd;
    if (hasNext) stage(kt + 32, p ^ 1);
    if (hasNext) { if (full) wait_async<3>(); else wait_async<1>(); }
    else         wait_async<0>();
    __syncthreads();

    union Frag { uint4 q[2]; v16bf v; };
    Frag af;
    const uint4* pa = (const uint4*)&lA[p][wave * 512 + lane * 16];
    af.q[0] = pa[0]; af.q[1] = pa[1];
#pragma unroll
    for (int nt = 0; nt < 4; ++nt) {
      Frag bfg;
      const uint4* pb = (const uint4*)&lB[p][nt * 512 + lane * 16];
      bfg.q[0] = pb[0]; bfg.q[1] = pb[1];
      acc[nt] = __builtin_amdgcn_wmma_f32_16x16x32_bf16(
          false, af.v, false, bfg.v, (short)0, acc[nt], false, false);
    }
    __syncthreads();
    p ^= 1;
  }

  // store D (f32 acc -> bf16). C/D layout: n = lane&15, VGPR r -> M = r + 8*(lane>=16)
  const int mhi = (lane >> 4) << 3;
#pragma unroll
  for (int nt = 0; nt < 4; ++nt) {
    int n = tN * 64 + nt * 16 + (lane & 15);
#pragma unroll
    for (int r = 0; r < 8; ++r) {
      int m = tM + wave * 16 + mhi + r;
      if (m < Mn) D[(size_t)m * Nc + n] = __float2bfloat16(acc[nt][r]);
    }
  }
}

// ---------------- per-(node,head) attention coefficients ----------------
__global__ __launch_bounds__(256) void node_attn_sd(
    const __hip_bfloat16* __restrict__ hb,
    const float* __restrict__ a_src, const float* __restrict__ a_dst,
    float* __restrict__ s, float* __restrict__ d) {
  int p = blockIdx.x * 8 + (threadIdx.x >> 5);
  if (p >= Mn * Hh) return;
  int lane = threadIdx.x & 31;
  int n = p / Hh, h = p - n * Hh;
  const __hip_bfloat16* hp = hb + (size_t)n * Nc + h * Cc;
  const float* as = a_src + h * Cc;
  const float* ad = a_dst + h * Cc;
  float ss = 0.f, dd = 0.f;
  for (int c = lane; c < Cc; c += 32) {
    float v = __bfloat162float(hp[c]);
    ss += v * as[c];
    dd += v * ad[c];
  }
#pragma unroll
  for (int off = 16; off >= 1; off >>= 1) {
    ss += __shfl_xor(ss, off, 32);
    dd += __shfl_xor(dd, off, 32);
  }
  if (lane == 0) { s[p] = ss; d[p] = dd; }
}

// ---------------- segment softmax over incoming edges ----------------
__device__ __forceinline__ void edge_ends(const int* ei, int e, int& src, int& dst) {
  if (e < Eg) { src = ei[e]; dst = ei[Eg + e]; }
  else        { src = e - Eg; dst = e - Eg; }     // self loop
}

__global__ void edge_max(const int* __restrict__ ei,
                         const float* __restrict__ s, const float* __restrict__ d,
                         unsigned* __restrict__ mkey) {
  int idx = blockIdx.x * blockDim.x + threadIdx.x;
  if (idx >= Et * Hh) return;
  int e = idx / Hh, h = idx - e * Hh;
  int src, dst; edge_ends(ei, e, src, dst);
  float v = s[src * Hh + h] + d[dst * Hh + h];
  v = (v > 0.f) ? v : 0.2f * v;                   // leaky_relu 0.2
  atomicMax(&mkey[dst * Hh + h], fkey(v));
}

__global__ void edge_sum(const int* __restrict__ ei,
                         const float* __restrict__ s, const float* __restrict__ d,
                         const unsigned* __restrict__ mkey,
                         float* __restrict__ den, float* __restrict__ exbuf) {
  int idx = blockIdx.x * blockDim.x + threadIdx.x;
  if (idx >= Et * Hh) return;
  int e = idx / Hh, h = idx - e * Hh;
  int src, dst; edge_ends(ei, e, src, dst);
  float v = s[src * Hh + h] + d[dst * Hh + h];
  v = (v > 0.f) ? v : 0.2f * v;
  float ex = __expf(v - fdec(mkey[dst * Hh + h]));
  exbuf[idx] = ex;
  atomicAdd(&den[dst * Hh + h], ex);
}

// one block per edge: agg[dst, c] += (1/H) * sum_h h[src,h,c] * alpha[e,h]
__global__ __launch_bounds__(256) void edge_scatter(
    const int* __restrict__ ei, const __hip_bfloat16* __restrict__ hb,
    const float* __restrict__ exbuf, const float* __restrict__ den,
    float* __restrict__ agg) {
  int e = blockIdx.x;
  int src, dst; edge_ends(ei, e, src, dst);
  __shared__ float sal[Hh];
  if (threadIdx.x < Hh)
    sal[threadIdx.x] = exbuf[e * Hh + threadIdx.x] /
                       (den[dst * Hh + threadIdx.x] + 1e-16f);
  __syncthreads();
  const __hip_bfloat16* hp = hb + (size_t)src * Nc;
  float* op = agg + (size_t)dst * Cc;
  for (int c = threadIdx.x; c < Cc; c += 256) {
    float a = 0.f;
#pragma unroll
    for (int h = 0; h < Hh; ++h)
      a += __bfloat162float(hp[h * Cc + c]) * sal[h];
    atomicAdd(&op[c], a * (1.0f / Hh));
  }
}

// ---------------- bias add + re-encode to bf16 (layer input) ----------------
__global__ void bias_to_bf16(const float* __restrict__ agg,
                             const float* __restrict__ b,
                             __hip_bfloat16* __restrict__ out) {
  int i = blockIdx.x * blockDim.x + threadIdx.x;
  if (i >= Mn * Cc) return;
  out[i] = __float2bfloat16(agg[i] + b[i % Cc]);
}

// ---------------- graph mean pool ----------------
__global__ void pool(const float* __restrict__ agg, const float* __restrict__ b,
                     const int* __restrict__ batch,
                     float* __restrict__ gsum, float* __restrict__ cnt) {
  int i = blockIdx.x * blockDim.x + threadIdx.x;
  if (i >= Mn * Cc) return;
  int n = i / Cc, c = i - n * Cc;
  int g = batch[n];
  atomicAdd(&gsum[g * Cc + c], agg[i] + b[c]);
  if (c == 0) atomicAdd(&cnt[g], 1.0f);
}

// ---------------- tiny MLP head ----------------
__global__ __launch_bounds__(128) void mlp_head(
    const float* __restrict__ gsum, const float* __restrict__ cnt,
    const float* __restrict__ w1, const float* __restrict__ b1,
    const float* __restrict__ w2, const float* __restrict__ b2,
    float* __restrict__ out) {
  int g = blockIdx.x;
  __shared__ float gs[Cc];
  __shared__ float z[128];
  float inv = 1.0f / fmaxf(cnt[g], 1.0f);
  for (int c = threadIdx.x; c < Cc; c += 128) gs[c] = gsum[g * Cc + c] * inv;
  __syncthreads();
  int j = threadIdx.x;
  float a = b1[j];
  for (int c = 0; c < Cc; ++c) a += gs[c] * w1[c * 128 + j];
  z[j] = fmaxf(a, 0.f);
  __syncthreads();
  if (j < 4) {
    float o = b2[j];
#pragma unroll
    for (int k = 0; k < 128; ++k) o += z[k] * w2[k * 4 + j];
    out[g * 4 + j] = o;
  }
}

// ---------------- launcher ----------------
extern "C" void kernel_launch(void* const* d_in, const int* in_sizes, int n_in,
                              void* d_out, int out_size, void* d_ws, size_t ws_size,
                              hipStream_t stream) {
  const float* x    = (const float*)d_in[0];
  const int*   ei   = (const int*)d_in[1];
  const int*   bat  = (const int*)d_in[2];
  const float* W1   = (const float*)d_in[3];
  const float* as1  = (const float*)d_in[4];
  const float* ad1  = (const float*)d_in[5];
  const float* b1   = (const float*)d_in[6];
  const float* W2   = (const float*)d_in[7];
  const float* as2  = (const float*)d_in[8];
  const float* ad2  = (const float*)d_in[9];
  const float* b2   = (const float*)d_in[10];
  const float* mw1  = (const float*)d_in[11];
  const float* mb1  = (const float*)d_in[12];
  const float* mw2  = (const float*)d_in[13];
  const float* mb2  = (const float*)d_in[14];
  float* out = (float*)d_out;

  char* ws = (char*)d_ws;
  size_t off = 0;
  auto carve = [&](size_t bytes) {
    void* p = ws + off;
    off = (off + bytes + 255) & ~(size_t)255;
    return p;
  };
  __hip_bfloat16* xb  = (__hip_bfloat16*)carve((size_t)Mn * Kd * 2);
  __hip_bfloat16* Wb  = (__hip_bfloat16*)carve((size_t)Kd * Nc * 2);
  __hip_bfloat16* hb  = (__hip_bfloat16*)carve((size_t)Mn * Nc * 2);
  float*    sArr = (float*)carve((size_t)Mn * Hh * 4);
  float*    dArr = (float*)carve((size_t)Mn * Hh * 4);
  unsigned* mkey = (unsigned*)carve((size_t)Mn * Hh * 4);
  float*    den  = (float*)carve((size_t)Mn * Hh * 4);
  float*    exb  = (float*)carve((size_t)Et * Hh * 4);
  float*    agg  = (float*)carve((size_t)Mn * Cc * 4);
  float*    gsum = (float*)carve((size_t)Gg * Cc * 4);
  float*    cnt  = (float*)carve((size_t)Gg * 4);

  const dim3 gemmGrid((Mn + 127) / 128, NTN);

  auto gat_layer = [&](const __hip_bfloat16* xin, const float* W,
                       const float* as, const float* ad) {
    w_to_bf16_swz<<<(Kd * Nc / 8 + 255) / 256, 256, 0, stream>>>(W, Wb);
    gemm_bf16_wmma<<<gemmGrid, 256, 0, stream>>>(xin, Wb, hb);
    node_attn_sd<<<(Mn * Hh + 7) / 8, 256, 0, stream>>>(hb, as, ad, sArr, dArr);
    zero_u32<<<512, 256, 0, stream>>>(mkey, (size_t)Mn * Hh);
    zero_u32<<<512, 256, 0, stream>>>((unsigned*)den, (size_t)Mn * Hh);
    zero_u32<<<2048, 256, 0, stream>>>((unsigned*)agg, (size_t)Mn * Cc);
    edge_max<<<(Et * Hh + 255) / 256, 256, 0, stream>>>(ei, sArr, dArr, mkey);
    edge_sum<<<(Et * Hh + 255) / 256, 256, 0, stream>>>(ei, sArr, dArr, mkey, den, exb);
    edge_scatter<<<Et, 256, 0, stream>>>(ei, hb, exb, den, agg);
  };

  // layer 1
  f32_to_bf16<<<2048, 256, 0, stream>>>(x, xb, Mn * Kd);
  gat_layer(xb, W1, as1, ad1);
  // layer 1 output (+b1) becomes layer 2 bf16 input
  bias_to_bf16<<<(Mn * Cc + 255) / 256, 256, 0, stream>>>(agg, b1, xb);
  // layer 2
  gat_layer(xb, W2, as2, ad2);
  // pool (+b2) and head
  zero_u32<<<256, 256, 0, stream>>>((unsigned*)gsum, (size_t)Gg * Cc);
  zero_u32<<<1, 64, 0, stream>>>((unsigned*)cnt, (size_t)Gg);
  pool<<<(Mn * Cc + 255) / 256, 256, 0, stream>>>(agg, b2, bat, gsum, cnt);
  mlp_head<<<Gg, 128, 0, stream>>>(gsum, cnt, mw1, mb1, mw2, mb2, out);
}
